// AttentionDecoder2D_10170482557368
// MI455X (gfx1250) — matrix operand order repaired and checked by reading, not compiled, not run on hardware
//
#include <hip/hip_runtime.h>
#include <hip/hip_bf16.h>

// ---------------------------------------------------------------------------
// Problem constants (from reference): B=64, T=32, H=512, V=512, C=64, S=8
// FLAT = 4096, NL = 2, HID = H*NL = 1024
// ---------------------------------------------------------------------------
#define BATCH 64
#define TSTEPS 32
#define HDIM 512
#define VOCAB 512
#define CHAN 64
#define FLATD 4096
#define HIDD 1024

typedef __attribute__((ext_vector_type(16))) __bf16 bf16x16;
typedef __attribute__((ext_vector_type(8)))  __bf16 bf16x8;
typedef __attribute__((ext_vector_type(8)))  float  f32x8;

// ---------------------------------------------------------------------------
// Generic wave-tiled bf16 GEMM:  D(MxN,f32) = A(MxK,bf16) @ BT(NxK,bf16)^T
//                                [+ bias(N)] [+ addend(MxN,f32)]
// One wave32 computes one 16x16 output tile with v_wmma_f32_16x16x32_bf16.
// BT is stored transposed (N rows of K contiguous bf16) so every lane does
// aligned contiguous b128 loads for both operands.
// ---------------------------------------------------------------------------
__global__ __launch_bounds__(256)
void gemm_bf16_tn(const __bf16* __restrict__ A, int lda,
                  const __bf16* __restrict__ BT, int ldb,
                  float* __restrict__ D, int ldd,
                  const float* __restrict__ bias,
                  const float* __restrict__ addend, int ldadd,
                  int M, int N, int K)
{
    const int lane = threadIdx.x & 31;
    const int wave = threadIdx.x >> 5;
    const int mtiles = M >> 4;
    const int ntiles = N >> 4;
    const int tile = blockIdx.x * 8 + wave;
    if (tile >= mtiles * ntiles) return;          // wave-uniform: EXEC stays all-1s for WMMA
    const int mt = tile % mtiles;
    const int nt = tile / mtiles;
    const int hl = lane >> 4;                     // half-wave selector (0 or 1)
    const int lm = lane & 15;
    const int row = (mt << 4) + lm;               // A row this lane feeds
    const int col = (nt << 4) + lm;               // B/D column this lane owns

    f32x8 acc;
#pragma unroll
    for (int r = 0; r < 8; ++r)
        acc[r] = addend ? addend[(size_t)((mt << 4) + r + (hl << 3)) * ldadd + col] : 0.0f;

    // A lane layout (16-bit 16x32): elems 0..7 = K[k0 + 8*hl .. +7],
    //                               elems 8..15 = K[k0+16+8*hl .. +7]
    const __bf16* ap = A  + (size_t)row * lda + (hl << 3);
    // B lane layout (32x16): same K order as A: elems 0..7 = K[k0 + 8*hl .. +7],
    //                       elems 8..15 = K[k0+16+8*hl .. +7] for column `col`
    const __bf16* bp = BT + (size_t)col * ldb + (hl << 3);

    for (int k0 = 0; k0 < K; k0 += 32) {
        bf16x8 a_lo = *(const bf16x8*)(ap);
        bf16x8 a_hi = *(const bf16x8*)(ap + 16);
        bf16x8 b_lo = *(const bf16x8*)(bp);
        bf16x8 b_hi = *(const bf16x8*)(bp + 16);
        __builtin_prefetch((const void*)(bp + 32), 0, 1);   // global_prefetch next K-slab of B
        ap += 32; bp += 32;
        bf16x16 a = __builtin_shufflevector(a_lo, a_hi, 0,1,2,3,4,5,6,7,8,9,10,11,12,13,14,15);
        bf16x16 b = __builtin_shufflevector(b_lo, b_hi, 0,1,2,3,4,5,6,7,8,9,10,11,12,13,14,15);
        acc = __builtin_amdgcn_wmma_f32_16x16x32_bf16(false, a, false, b,
                                                      (short)0, acc, false, false);
    }

    // C/D layout: VGPR r, lanes<16 -> M = 16*mt + r, lanes>=16 -> M = 16*mt + r + 8; N = col
#pragma unroll
    for (int r = 0; r < 8; ++r) {
        const int m = (mt << 4) + r + (hl << 3);
        float v = acc[r];
        if (bias) v += bias[col];
        D[(size_t)m * ldd + col] = v;
    }
}

// ---------------------------------------------------------------------------
// Setup kernels: fp32 -> bf16 conversion / transposition of weights
// ---------------------------------------------------------------------------
__global__ void f32_to_bf16_kernel(const float* __restrict__ src, __bf16* __restrict__ dst, int n)
{
    int i = blockIdx.x * blockDim.x + threadIdx.x;
    if (i < n) dst[i] = (__bf16)src[i];
}

// W (K x N) row-major fp32  ->  WT (N x K) row-major bf16
__global__ void transpose_bf16_kernel(const float* __restrict__ W, __bf16* __restrict__ WT,
                                      int K, int N)
{
    int i = blockIdx.x * blockDim.x + threadIdx.x;
    if (i >= K * N) return;
    int k = i % K, n = i / K;                      // coalesced writes along K
    WT[(size_t)n * K + k] = (__bf16)W[(size_t)k * N + n];
}

// [W0;W1;W2] stacked along K, each (Ki x N) -> WT (N x (K0+K1+K2)) bf16
__global__ void cat3_transpose_bf16_kernel(const float* __restrict__ W0,
                                           const float* __restrict__ W1,
                                           const float* __restrict__ W2,
                                           int K0, int K1, int K2, int N,
                                           __bf16* __restrict__ WT)
{
    const int Kt = K0 + K1 + K2;
    int i = blockIdx.x * blockDim.x + threadIdx.x;
    if (i >= Kt * N) return;
    int k = i % Kt, n = i / Kt;
    float v;
    if (k < K0)            v = W0[(size_t)k * N + n];
    else if (k < K0 + K1)  v = W1[(size_t)(k - K0) * N + n];
    else                   v = W2[(size_t)(k - K0 - K1) * N + n];
    WT[(size_t)n * Kt + k] = (__bf16)v;
}

// zero recurrent state + combined layer-1 bias (re-done every launch: deterministic)
__global__ void init_state_kernel(__bf16* hidbf, __bf16* xcat1, __bf16* xcat0,
                                  float* c0, float* c1, float* bsum,
                                  const float* __restrict__ bih1,
                                  const float* __restrict__ bhh1)
{
    int i = blockIdx.x * blockDim.x + threadIdx.x;   // 65536 threads
    if (i < BATCH * HIDD) { hidbf[i] = (__bf16)0.0f; xcat1[i] = (__bf16)0.0f; }
    if (i < BATCH * HDIM) {
        c0[i] = 0.0f; c1[i] = 0.0f;
        int b = i >> 9, n = i & 511;
        xcat0[b * 1536 + 512 + n] = (__bf16)0.0f;    // h0 slot of [emb|h0|att]
    }
    if (i < 4 * HDIM) bsum[i] = bih1[i] + bhh1[i];
}

// ---------------------------------------------------------------------------
// Softmax over 4096 attention logits, visual context, attended projection,
// and embedding gather into Xcat0 = [emb | h0 | attended].  One block / batch.
// ---------------------------------------------------------------------------
__global__ __launch_bounds__(256)
void attn_softmax_ctx_kernel(const float* __restrict__ logits,   // (64,4096)
                             const float* __restrict__ vflat,    // (64,4096)
                             const float* __restrict__ aow,      // (64,512)
                             const float* __restrict__ aob,      // (512)
                             const __bf16* __restrict__ embbf,   // (512,512)
                             const int*   __restrict__ caption,  // (64,32)
                             int tstep,
                             __bf16* __restrict__ xcat0)         // (64,1536)
{
    __shared__ float red[256];
    __shared__ float chan[CHAN];
    const int b = blockIdx.x;
    const int t = threadIdx.x;
    const float* lrow = logits + (size_t)b * FLATD;
    const float* vrow = vflat  + (size_t)b * FLATD;

    // max reduction (each thread owns 16 contiguous elements, all in one channel)
    float mx = -3.4e38f;
#pragma unroll
    for (int j = 0; j < 16; ++j) mx = fmaxf(mx, lrow[t * 16 + j]);
    red[t] = mx; __syncthreads();
    for (int s = 128; s > 0; s >>= 1) { if (t < s) red[t] = fmaxf(red[t], red[t + s]); __syncthreads(); }
    mx = red[0]; __syncthreads();

    // exp, partition sum, per-channel weighted visual sum
    float zs = 0.0f, pv = 0.0f;
#pragma unroll
    for (int j = 0; j < 16; ++j) {
        float e = __expf(lrow[t * 16 + j] - mx);
        zs += e;
        pv += e * vrow[t * 16 + j];
    }
    red[t] = zs; __syncthreads();
    for (int s = 128; s > 0; s >>= 1) { if (t < s) red[t] += red[t + s]; __syncthreads(); }
    const float Z = red[0]; __syncthreads();

    red[t] = pv; __syncthreads();
    if (t < CHAN)   // 4 threads per channel of 64 spatial positions
        chan[t] = (red[t * 4] + red[t * 4 + 1] + red[t * 4 + 2] + red[t * 4 + 3]) / Z;
    __syncthreads();

    // attended = applied @ attn_out_w + attn_out_b ;  emb gather
    const int tok = caption[b * TSTEPS + tstep];
    __bf16* xrow = xcat0 + (size_t)b * 1536;
    for (int n = t; n < HDIM; n += 256) {
        float s = aob[n];
#pragma unroll 8
        for (int c = 0; c < CHAN; ++c) s += chan[c] * aow[c * HDIM + n];
        xrow[1024 + n] = (__bf16)s;                 // attended slot
        xrow[n] = embbf[(size_t)tok * HDIM + n];    // emb slot
    }
}

// ---------------------------------------------------------------------------
// Pointwise LSTM updates (fp32 state, bf16 h broadcast into next GEMM inputs)
// ---------------------------------------------------------------------------
__device__ __forceinline__ float sigmoid_f(float x) { return 1.0f / (1.0f + __expf(-x)); }

__global__ void lstm0_pw_kernel(const float* __restrict__ g, float* __restrict__ c0,
                                __bf16* hidbf, __bf16* xcat0, __bf16* xcat1)
{
    int i = blockIdx.x * blockDim.x + threadIdx.x;   // 64*512
    int b = i >> 9, n = i & 511;
    const float* gr = g + (size_t)b * 2048;
    float ig = sigmoid_f(gr[n]);
    float fg = sigmoid_f(gr[512 + n]);
    float cg = tanhf(gr[1024 + n]);
    float og = sigmoid_f(gr[1536 + n]);
    float c = fg * c0[i] + ig * cg;
    c0[i] = c;
    __bf16 hb = (__bf16)(og * tanhf(c));
    hidbf[b * HIDD + n]       = hb;                  // attn-GEMM A, h0 half
    xcat0[b * 1536 + 512 + n] = hb;                  // gates0 A, h0 slot
    xcat1[b * HIDD + n]       = hb;                  // gates1 A, h0 slot
}

__global__ void lstm1_pw_kernel(const float* __restrict__ g, float* __restrict__ c1,
                                __bf16* hidbf, __bf16* xcat1)
{
    int i = blockIdx.x * blockDim.x + threadIdx.x;
    int b = i >> 9, n = i & 511;
    const float* gr = g + (size_t)b * 2048;
    float ig = sigmoid_f(gr[n]);
    float fg = sigmoid_f(gr[512 + n]);
    float cg = tanhf(gr[1024 + n]);
    float og = sigmoid_f(gr[1536 + n]);
    float c = fg * c1[i] + ig * cg;
    c1[i] = c;
    __bf16 hb = (__bf16)(og * tanhf(c));
    hidbf[b * HIDD + 512 + n] = hb;                  // attn-GEMM A, h1 half
    xcat1[b * HIDD + 512 + n] = hb;                  // gates1 A / vocab-GEMM A
}

// ---------------------------------------------------------------------------
// Host-side orchestration
// ---------------------------------------------------------------------------
extern "C" void kernel_launch(void* const* d_in, const int* in_sizes, int n_in,
                              void* d_out, int out_size, void* d_ws, size_t ws_size,
                              hipStream_t stream)
{
    const float* visual    = (const float*)d_in[0];   // (64,64,8,8) == vflat (64,4096)
    const int*   caption   = (const int*)  d_in[1];   // (64,32)
    const float* embedding = (const float*)d_in[2];   // (512,512)
    const float* attn_w    = (const float*)d_in[3];   // (5120,4096)
    const float* attn_b    = (const float*)d_in[4];   // (4096)
    const float* aow       = (const float*)d_in[5];   // (64,512)
    const float* aob       = (const float*)d_in[6];   // (512)
    const float* w0        = (const float*)d_in[7];   // (512,2048)
    const float* u0        = (const float*)d_in[8];
    const float* z0        = (const float*)d_in[9];
    const float* b0        = (const float*)d_in[10];  // (2048)
    const float* wih1      = (const float*)d_in[11];
    const float* whh1      = (const float*)d_in[12];
    const float* bih1      = (const float*)d_in[13];
    const float* bhh1      = (const float*)d_in[14];
    const float* out_w     = (const float*)d_in[15];  // (512,512)
    const float* out_b     = (const float*)d_in[16];  // (512)
    float* out = (float*)d_out;                       // (64,32,512)

    char*  ws  = (char*)d_ws;
    size_t off = 0;
    auto alloc = [&](size_t bytes) -> void* {
        void* p = ws + off;
        off += (bytes + 255) & ~(size_t)255;
        return p;
    };

    __bf16* attnWT2 = (__bf16*)alloc((size_t)FLATD * HIDD  * 2);  // attn_w[4096:]^T  (4096,1024)
    __bf16* wcat0T  = (__bf16*)alloc((size_t)2048 * 1536   * 2);  // [w0;u0;z0]^T     (2048,1536)
    __bf16* wcat1T  = (__bf16*)alloc((size_t)2048 * 1024   * 2);  // [wih1;whh1]^T    (2048,1024)
    __bf16* outWT   = (__bf16*)alloc((size_t)VOCAB * HDIM  * 2);  // out_w^T          (512,512)
    __bf16* embbf   = (__bf16*)alloc((size_t)VOCAB * HDIM  * 2);  // embedding bf16
    __bf16* vflatbf = (__bf16*)alloc((size_t)BATCH * FLATD * 2);  // visual bf16
    float*  vbase   = (float*) alloc((size_t)BATCH * FLATD * 4);  // hoisted vflat@Wv + b
    float*  logitsA = (float*) alloc((size_t)BATCH * FLATD * 4);  // per-step attn logits
    __bf16* hidbf   = (__bf16*)alloc((size_t)BATCH * HIDD  * 2);  // [h0|h1] bf16
    __bf16* xcat0   = (__bf16*)alloc((size_t)BATCH * 1536  * 2);  // [emb|h0|att] bf16
    __bf16* xcat1   = (__bf16*)alloc((size_t)BATCH * HIDD  * 2);  // [h0|h1] bf16
    float*  g0buf   = (float*) alloc((size_t)BATCH * 2048  * 4);
    float*  g1buf   = (float*) alloc((size_t)BATCH * 2048  * 4);
    float*  c0buf   = (float*) alloc((size_t)BATCH * HDIM  * 4);
    float*  c1buf   = (float*) alloc((size_t)BATCH * HDIM  * 4);
    float*  bsum1   = (float*) alloc((size_t)2048 * 4);
    __bf16* attnWT1 = (__bf16*)alloc((size_t)FLATD * FLATD * 2);  // attn_w[:4096]^T (setup only)

    // ---- setup: weight conversion + hoisted visual GEMM -------------------
    transpose_bf16_kernel<<<dim3((FLATD * FLATD) / 256), dim3(256), 0, stream>>>(
        attn_w, attnWT1, FLATD, FLATD);
    transpose_bf16_kernel<<<dim3((HIDD * FLATD) / 256), dim3(256), 0, stream>>>(
        attn_w + (size_t)FLATD * FLATD, attnWT2, HIDD, FLATD);
    cat3_transpose_bf16_kernel<<<dim3((1536 * 2048) / 256), dim3(256), 0, stream>>>(
        w0, u0, z0, 512, 512, 512, 2048, wcat0T);
    cat3_transpose_bf16_kernel<<<dim3((1024 * 2048) / 256), dim3(256), 0, stream>>>(
        wih1, whh1, wih1, 512, 512, 0, 2048, wcat1T);
    transpose_bf16_kernel<<<dim3((HDIM * VOCAB) / 256), dim3(256), 0, stream>>>(
        out_w, outWT, HDIM, VOCAB);
    f32_to_bf16_kernel<<<dim3((VOCAB * HDIM) / 256), dim3(256), 0, stream>>>(
        embedding, embbf, VOCAB * HDIM);
    f32_to_bf16_kernel<<<dim3((BATCH * FLATD) / 256), dim3(256), 0, stream>>>(
        visual, vflatbf, BATCH * FLATD);
    init_state_kernel<<<dim3(256), dim3(256), 0, stream>>>(
        hidbf, xcat1, xcat0, c0buf, c1buf, bsum1, bih1, bhh1);

    // vbase = vflat @ attn_w[:4096,:] + attn_b   (64x4096x4096, once)
    gemm_bf16_tn<<<dim3(128), dim3(256), 0, stream>>>(
        vflatbf, FLATD, attnWT1, FLATD, vbase, FLATD,
        attn_b, nullptr, 0, BATCH, FLATD, FLATD);

    // ---- sequential decode loop ------------------------------------------
    for (int t = 0; t < TSTEPS; ++t) {
        // attn logits = vbase + [h0|h1] @ attn_w[4096:,:]   (64x4096, K=1024)
        gemm_bf16_tn<<<dim3(128), dim3(256), 0, stream>>>(
            hidbf, HIDD, attnWT2, HIDD, logitsA, FLATD,
            nullptr, vbase, FLATD, BATCH, FLATD, HIDD);

        // softmax -> applied -> attended ; emb gather -> Xcat0
        attn_softmax_ctx_kernel<<<dim3(BATCH), dim3(256), 0, stream>>>(
            logitsA, visual, aow, aob, embbf, caption, t, xcat0);

        // g0 = [emb|h0|att] @ [w0;u0;z0] + b0   (64x2048, K=1536)
        gemm_bf16_tn<<<dim3(64), dim3(256), 0, stream>>>(
            xcat0, 1536, wcat0T, 1536, g0buf, 2048,
            b0, nullptr, 0, BATCH, 2048, 1536);

        lstm0_pw_kernel<<<dim3(128), dim3(256), 0, stream>>>(
            g0buf, c0buf, hidbf, xcat0, xcat1);

        // g1 = [h0|h1] @ [wih1;whh1] + (bih1+bhh1)   (64x2048, K=1024)
        gemm_bf16_tn<<<dim3(64), dim3(256), 0, stream>>>(
            xcat1, HIDD, wcat1T, HIDD, g1buf, 2048,
            bsum1, nullptr, 0, BATCH, 2048, 1024);

        lstm1_pw_kernel<<<dim3(128), dim3(256), 0, stream>>>(
            g1buf, c1buf, hidbf, xcat1);

        // logits[:,t,:] = h1 @ out_w + out_b   (64x512, K=512) straight to d_out
        gemm_bf16_tn<<<dim3(16), dim3(256), 0, stream>>>(
            xcat1 + 512, HIDD, outWT, HDIM, out + (size_t)t * VOCAB, TSTEPS * VOCAB,
            out_b, nullptr, 0, BATCH, VOCAB, HDIM);
    }
}
